// TransformerLayer_91336774517206
// MI455X (gfx1250) — compile-verified, matrix operand order checked by reading, not run on hardware
//
#include <hip/hip_runtime.h>
#include <math.h>

// ---------------------------------------------------------------------------
// Types for CDNA5 WMMA (wave32): v_wmma_f32_16x16x32_bf16
// ---------------------------------------------------------------------------
typedef __bf16 bf16x16 __attribute__((ext_vector_type(16)));
typedef __bf16 bf16x8  __attribute__((ext_vector_type(8)));
typedef float  f32x8   __attribute__((ext_vector_type(8)));
typedef unsigned int u32x4 __attribute__((ext_vector_type(4)));
typedef int i32x4 __attribute__((ext_vector_type(4)));
typedef int i32x8 __attribute__((ext_vector_type(8)));

union Frag {
    bf16x16 v;
    bf16x8  h[2];
};

__device__ __forceinline__ __bf16 to_bf16(float f) {
    unsigned u = __builtin_bit_cast(unsigned, f);
    unsigned r = u + 0x7FFFu + ((u >> 16) & 1u);   // round-to-nearest-even
    unsigned short hs = (unsigned short)(r >> 16);
    return __builtin_bit_cast(__bf16, hs);
}

__device__ __forceinline__ f32x8 wmma_bf16(const Frag& a, const Frag& b, f32x8 c) {
    return __builtin_amdgcn_wmma_f32_16x16x32_bf16(
        /*neg_a=*/false, a.v, /*neg_b=*/false, b.v,
        /*c_mod=*/(short)0, c, /*reuse_a=*/false, /*reuse_b=*/false);
}

// ---------------------------------------------------------------------------
// Tensor Data Mover (gfx1250): 2D tile load global -> LDS via D# descriptor.
// D# layout per CDNA5 ISA ch.8. Issued by one wave; waited with TENSORcnt.
// ---------------------------------------------------------------------------
#if __has_builtin(__builtin_amdgcn_tensor_load_to_lds) && __has_builtin(__builtin_amdgcn_s_wait_tensorcnt)
#define HAVE_TDM 1
#else
#define HAVE_TDM 0
#endif

#if HAVE_TDM
__device__ __forceinline__ void tdm_load_2d(unsigned lds_off, const void* gptr,
                                            unsigned tile_w,   // elements (2B)
                                            unsigned tile_h,   // rows
                                            unsigned long long row_stride /*elements*/) {
    unsigned long long ga = (unsigned long long)(size_t)gptr;
    u32x4 g0;
    g0[0] = 1u;                                             // count=1, user mode
    g0[1] = lds_off;                                        // lds_addr (bytes)
    g0[2] = (unsigned)(ga & 0xFFFFFFFFu);                   // global_addr lo
    g0[3] = (unsigned)((ga >> 32) & 0x01FFFFFFu) | (2u << 30); // addr hi | type=2
    i32x8 g1;
    g1[0] = (int)(1u << 16);                                // data_size=1 (2 bytes)
    g1[1] = (int)((tile_w & 0xFFFFu) << 16);                // tensor_dim0 lo
    g1[2] = (int)(((tile_w >> 16) & 0xFFFFu) | ((tile_h & 0xFFFFu) << 16)); // dim0 hi | tensor_dim1 lo
    g1[3] = (int)(((tile_h >> 16) & 0xFFFFu) | ((tile_w & 0xFFFFu) << 16)); // dim1 hi | tile_dim0
    g1[4] = (int)(tile_h & 0xFFFFu);                        // tile_dim1 | tile_dim2=0
    g1[5] = (int)(row_stride & 0xFFFFFFFFull);              // tensor_dim0_stride lo
    g1[6] = (int)((row_stride >> 32) & 0xFFFFull);          // stride hi | dim1_stride lo=0
    g1[7] = 0;
    i32x4 gz = {0, 0, 0, 0};
#if __has_include(<hip/amd_detail/amd_gfx1250_TDM.h>)
    i32x8 gz8 = {0, 0, 0, 0, 0, 0, 0, 0};
    __builtin_amdgcn_tensor_load_to_lds(g0, g1, gz, gz, gz8, 0);   // clang-23 6-arg form
#else
    __builtin_amdgcn_tensor_load_to_lds(g0, g1, gz, gz, 0);        // ROCm 7.2 5-arg form
#endif
}
#endif

// Problem constants (match reference)
#define BB   2
#define SS   2048
#define DD   2048
#define HH   16
#define HDD  128
#define FFD  8192
#define MM   (BB * SS)      // 4096 rows

// ---------------------------------------------------------------------------
// fp32 -> bf16 elementwise convert (weights)
// ---------------------------------------------------------------------------
__global__ __launch_bounds__(256)
void cvt_f32_bf16_kernel(const float* __restrict__ in, __bf16* __restrict__ out, size_t n4) {
    size_t i = (size_t)blockIdx.x * 256 + threadIdx.x;
    if (i >= n4) return;
    const float4 v = ((const float4*)in)[i];
    __bf16* o = out + 4 * i;
    o[0] = to_bf16(v.x); o[1] = to_bf16(v.y); o[2] = to_bf16(v.z); o[3] = to_bf16(v.w);
}

// ---------------------------------------------------------------------------
// LayerNorm (one block per row, D=2048), bf16 output
// ---------------------------------------------------------------------------
__global__ __launch_bounds__(256)
void layernorm_bf16_kernel(const float* __restrict__ x, const float* __restrict__ g,
                           const float* __restrict__ b, __bf16* __restrict__ out) {
    __shared__ float red[16];
    const int row = blockIdx.x;
    const float* xr = x + (size_t)row * DD;
    float s = 0.f, s2 = 0.f;
    for (int i = threadIdx.x; i < DD; i += 256) {
        float v = xr[i];
        s += v; s2 += v * v;
    }
    #pragma unroll
    for (int off = 16; off > 0; off >>= 1) {
        s  += __shfl_xor(s,  off, 32);
        s2 += __shfl_xor(s2, off, 32);
    }
    if ((threadIdx.x & 31) == 0) {
        red[threadIdx.x >> 5]       = s;
        red[8 + (threadIdx.x >> 5)] = s2;
    }
    __syncthreads();
    float ts = 0.f, ts2 = 0.f;
    #pragma unroll
    for (int w = 0; w < 8; ++w) { ts += red[w]; ts2 += red[8 + w]; }
    const float mu  = ts * (1.0f / DD);
    const float var = ts2 * (1.0f / DD) - mu * mu;
    const float inv = rsqrtf(var + 1e-5f);
    __bf16* orow = out + (size_t)row * DD;
    for (int i = threadIdx.x; i < DD; i += 256) {
        float v = (xr[i] - mu) * inv * g[i] + b[i];
        orow[i] = to_bf16(v);
    }
}

// ---------------------------------------------------------------------------
// Tiled bf16 WMMA GEMM: C[M,N] = A[M,K] @ B[K,N] (+bias, gelu, residual)
// Block tile 128x128, BK=64, 8 waves (2x4), each wave 64x32 (4x2 WMMA tiles).
// Double-buffered LDS; A tile staged by TDM (async, TENSORcnt) when available,
// so the DMA for tile i+1 runs underneath the 16 WMMAs of tile i.
// ---------------------------------------------------------------------------
template <bool BIAS, bool GELU, bool RES, bool STF32, bool STBF16>
__global__ __launch_bounds__(256)
void gemm_bf16_kernel(const __bf16* __restrict__ A, const __bf16* __restrict__ Bw,
                      const float* __restrict__ bias, const float* __restrict__ res,
                      float* __restrict__ outF, __bf16* __restrict__ outB,
                      int M, int N, int K) {
    __shared__ __align__(16) __bf16 sA[2][128 * 64];   // row-major [m][k]
    __shared__ __align__(16) __bf16 sB[2][128 * 64];   // transposed  [n][k]

    const int tid  = threadIdx.x;
    const int wave = tid >> 5;
    const int lane = tid & 31;
    const int half = lane >> 4;
    const int l16  = lane & 15;
    const int wm0  = (wave >> 2) * 64;
    const int wn0  = (wave & 3) * 32;
    const int bm   = blockIdx.y * 128;
    const int bn   = blockIdx.x * 128;
    const int c0   = half * 8;

    f32x8 acc[4][2] = {};

    const int ar = tid >> 1, ac = (tid & 1) * 32;   // manual A staging: 128 x 64
    const int br = tid >> 2, bc = (tid & 3) * 32;   // B staging: 64 x 128

    auto stageA = [&](int buf, int k0) {
#if HAVE_TDM
        if (tid < 32)
            tdm_load_2d((unsigned)(size_t)&sA[buf][0], A + (size_t)bm * K + k0,
                        64, 128, (unsigned long long)K);
#else
        const __bf16* ag = A + (size_t)(bm + ar) * K + (k0 + ac);
        #pragma unroll
        for (int i = 0; i < 4; ++i)
            *(uint4*)(&sA[buf][ar * 64 + ac + 8 * i]) = *(const uint4*)(ag + 8 * i);
#endif
    };
    auto stageB = [&](int buf, int k0) {
        const __bf16* bg = Bw + (size_t)(k0 + br) * N + (bn + bc);
        __align__(16) __bf16 tmp[32];
        #pragma unroll
        for (int i = 0; i < 4; ++i)
            *(uint4*)(tmp + 8 * i) = *(const uint4*)(bg + 8 * i);
        #pragma unroll
        for (int i = 0; i < 32; ++i) sB[buf][(bc + i) * 64 + br] = tmp[i];
        if (k0 + 64 < K)                                 // gfx1250 global_prefetch_b8
            __builtin_prefetch(bg + (size_t)64 * N, 0, 1);
    };

    // ---- prologue: stage tile 0 ----
    stageA(0, 0);
    stageB(0, 0);
#if HAVE_TDM
    if (tid < 32) __builtin_amdgcn_s_wait_tensorcnt(0);
#endif
    __syncthreads();

    int cur = 0;
    for (int k0 = 0; k0 < K; k0 += 64) {
        const int nk = k0 + 64;
        // ---- kick off next tile (async TDM + manual B) into spare buffer ----
        if (nk < K) {
            stageA(cur ^ 1, nk);
            stageB(cur ^ 1, nk);
        }

        // ---- compute current tile: 2 k-steps of 32 ----
        #pragma unroll
        for (int kk = 0; kk < 64; kk += 32) {
            Frag af[4], bf[2];
            #pragma unroll
            for (int r = 0; r < 4; ++r) {
                const __bf16* p = &sA[cur][(wm0 + 16 * r + l16) * 64 + kk + c0];
                af[r].h[0] = *(const bf16x8*)(p);
                af[r].h[1] = *(const bf16x8*)(p + 16);
            }
            #pragma unroll
            for (int c = 0; c < 2; ++c) {
                const __bf16* p = &sB[cur][(wn0 + 16 * c + l16) * 64 + kk + c0];
                bf[c].h[0] = *(const bf16x8*)(p);
                bf[c].h[1] = *(const bf16x8*)(p + 16);
            }
            #pragma unroll
            for (int r = 0; r < 4; ++r)
                #pragma unroll
                for (int c = 0; c < 2; ++c)
                    acc[r][c] = wmma_bf16(af[r], bf[c], acc[r][c]);
        }

#if HAVE_TDM
        if (nk < K && tid < 32) __builtin_amdgcn_s_wait_tensorcnt(0);
#endif
        __syncthreads();      // publish next buffer, retire reads of current
        cur ^= 1;
    }

    // ---- epilogue (C layout: VGPR e -> row 16r + half*8 + e, col = 16c + l16) ----
    #pragma unroll
    for (int r = 0; r < 4; ++r) {
        #pragma unroll
        for (int c = 0; c < 2; ++c) {
            const int gcol = bn + wn0 + 16 * c + l16;
            const float bv = BIAS ? bias[gcol] : 0.f;
            #pragma unroll
            for (int e = 0; e < 8; ++e) {
                const int gm = bm + wm0 + 16 * r + half * 8 + e;
                float v = acc[r][c][e];
                if (BIAS) v += bv;
                if (GELU) v = 0.5f * v * (1.f + erff(v * 0.70710678118654752f));
                const size_t idx = (size_t)gm * N + gcol;
                if (RES)    v += res[idx];
                if (STF32)  outF[idx] = v;
                if (STBF16) outB[idx] = to_bf16(v);
            }
        }
    }
}

// ---------------------------------------------------------------------------
// Flash attention (causal, scale 1/sqrt(128)); qkv packed [M, 3D] bf16.
// Grid: (S/64 q-tiles, B*H). Block: 128 threads = 4 waves, 16 q-rows/wave.
// K tile staged by TDM when available; V transposed manually.
// ---------------------------------------------------------------------------
__global__ __launch_bounds__(128)
void flash_attn_kernel(const __bf16* __restrict__ qkv, __bf16* __restrict__ attn_out) {
    __shared__ __align__(16) __bf16 sK[64 * HDD];      // [key][d]
    __shared__ __align__(16) __bf16 sV[HDD * 64];      // transposed [d][key]
    __shared__ __align__(16) __bf16 sP[4][16 * 64];    // per-wave P tile

    const int bh    = blockIdx.y;
    const int b     = bh >> 4;
    const int h     = bh & 15;
    const int qtb   = blockIdx.x;
    const int qbase = qtb * 64;
    const int tid   = threadIdx.x;
    const int wave  = tid >> 5;
    const int lane  = tid & 31;
    const int half  = lane >> 4;
    const int l16   = lane & 15;
    const int c0    = half * 8;
    const size_t rs = 3 * DD;

    const __bf16* Qg = qkv + (size_t)b * SS * rs + (size_t)h * HDD;
    const __bf16* Kg = Qg + DD;
    const __bf16* Vg = Qg + 2 * DD;

    // Q fragments in registers (16 rows per wave, 4 k-steps over HD=128)
    Frag qf[4];
    {
        const int qrow = qbase + 16 * wave + l16;
        const __bf16* qp = Qg + (size_t)qrow * rs + c0;
        #pragma unroll
        for (int ks = 0; ks < 4; ++ks) {
            qf[ks].h[0] = *(const bf16x8*)(qp + 32 * ks);
            qf[ks].h[1] = *(const bf16x8*)(qp + 32 * ks + 16);
        }
    }

    f32x8 o[8] = {};
    float mrow[8], lrow[8];
    #pragma unroll
    for (int i = 0; i < 8; ++i) { mrow[i] = -3.0e38f; lrow[i] = 0.f; }
    const float scale = 0.088388347648318447f;   // 1/sqrt(128)

    const int ktiles = qtb + 1;                  // causal
    for (int kt = 0; kt < ktiles; ++kt) {
        const int kbase = kt * 64;
        // ---- stage K tile ----
#if HAVE_TDM
        if (tid < 32)
            tdm_load_2d((unsigned)(size_t)sK, Kg + (size_t)kbase * rs, HDD, 64,
                        (unsigned long long)rs);
#else
        {
            const int r = tid >> 1, cseg = (tid & 1) * 64;
            const __bf16* kp = Kg + (size_t)(kbase + r) * rs + cseg;
            #pragma unroll
            for (int i = 0; i < 8; ++i)
                *(uint4*)(&sK[r * HDD + cseg + 8 * i]) = *(const uint4*)(kp + 8 * i);
        }
#endif
        // ---- stage V tile transposed ----
        {
            const int r = tid >> 1, cseg = (tid & 1) * 64;
            const __bf16* vp = Vg + (size_t)(kbase + r) * rs + cseg;
            #pragma unroll
            for (int i = 0; i < 8; ++i) {
                __align__(16) __bf16 t[8];
                *(uint4*)t = *(const uint4*)(vp + 8 * i);
                #pragma unroll
                for (int j = 0; j < 8; ++j) sV[(cseg + 8 * i + j) * 64 + r] = t[j];
            }
        }
#if HAVE_TDM
        if (tid < 32) __builtin_amdgcn_s_wait_tensorcnt(0);
#endif
        __syncthreads();

        // ---- S = Q K^T : 16x64 per wave (batched fragment loads) ----
        f32x8 sc[4] = {};
        #pragma unroll
        for (int ct = 0; ct < 4; ++ct) {
            Frag kf[4];
            #pragma unroll
            for (int ks = 0; ks < 4; ++ks) {
                const __bf16* p = &sK[(16 * ct + l16) * HDD + 32 * ks + c0];
                kf[ks].h[0] = *(const bf16x8*)(p);
                kf[ks].h[1] = *(const bf16x8*)(p + 16);
            }
            #pragma unroll
            for (int ks = 0; ks < 4; ++ks)
                sc[ct] = wmma_bf16(qf[ks], kf[ks], sc[ct]);
        }

        // ---- scale + causal mask ----
        const int qrow0 = qbase + 16 * wave + half * 8;   // + e
        #pragma unroll
        for (int ct = 0; ct < 4; ++ct) {
            const int kcol = kbase + 16 * ct + l16;
            #pragma unroll
            for (int e = 0; e < 8; ++e) {
                float v = sc[ct][e] * scale;
                if (kcol > qrow0 + e) v = -3.0e38f;
                sc[ct][e] = v;
            }
        }

        // ---- online softmax (row = half*8+e lives in a 16-lane group) ----
        #pragma unroll
        for (int e = 0; e < 8; ++e) {
            float mx = fmaxf(fmaxf(sc[0][e], sc[1][e]), fmaxf(sc[2][e], sc[3][e]));
            #pragma unroll
            for (int off = 8; off > 0; off >>= 1) mx = fmaxf(mx, __shfl_xor(mx, off, 16));
            const float mnew = fmaxf(mrow[e], mx);
            const float alpha = __expf(mrow[e] - mnew);
            mrow[e] = mnew;
            lrow[e] *= alpha;
            #pragma unroll
            for (int t = 0; t < 8; ++t) o[t][e] *= alpha;
            float rsum = 0.f;
            #pragma unroll
            for (int ct = 0; ct < 4; ++ct) {
                float p = __expf(sc[ct][e] - mnew);
                sc[ct][e] = p;
                rsum += p;
            }
            #pragma unroll
            for (int off = 8; off > 0; off >>= 1) rsum += __shfl_xor(rsum, off, 16);
            lrow[e] += rsum;
        }

        // ---- repack P into A layout via per-wave LDS ----
        #pragma unroll
        for (int ct = 0; ct < 4; ++ct)
            #pragma unroll
            for (int e = 0; e < 8; ++e)
                sP[wave][(half * 8 + e) * 64 + 16 * ct + l16] = to_bf16(sc[ct][e]);

        // ---- O += P V (K=64 -> 2 k-steps; N=128 -> 8 col tiles, batched) ----
        #pragma unroll
        for (int ks = 0; ks < 2; ++ks) {
            Frag pf;
            const __bf16* pp = &sP[wave][l16 * 64 + 32 * ks + c0];
            pf.h[0] = *(const bf16x8*)(pp);
            pf.h[1] = *(const bf16x8*)(pp + 16);
            #pragma unroll
            for (int tb = 0; tb < 2; ++tb) {
                Frag vf[4];
                #pragma unroll
                for (int t4 = 0; t4 < 4; ++t4) {
                    const __bf16* vv = &sV[(16 * (tb * 4 + t4) + l16) * 64 + 32 * ks + c0];
                    vf[t4].h[0] = *(const bf16x8*)(vv);
                    vf[t4].h[1] = *(const bf16x8*)(vv + 16);
                }
                #pragma unroll
                for (int t4 = 0; t4 < 4; ++t4)
                    o[tb * 4 + t4] = wmma_bf16(pf, vf[t4], o[tb * 4 + t4]);
            }
        }
        __syncthreads();
    }

    // ---- normalize and store [B,S,D] bf16 ----
    __bf16* Og = attn_out + (size_t)b * SS * DD + (size_t)h * HDD;
    #pragma unroll
    for (int t = 0; t < 8; ++t) {
        const int d = 16 * t + l16;
        #pragma unroll
        for (int e = 0; e < 8; ++e) {
            const int srow = qbase + 16 * wave + half * 8 + e;
            Og[(size_t)srow * DD + d] = to_bf16(o[t][e] / lrow[e]);
        }
    }
}

// ---------------------------------------------------------------------------
// Launcher
// ---------------------------------------------------------------------------
extern "C" void kernel_launch(void* const* d_in, const int* in_sizes, int n_in,
                              void* d_out, int out_size, void* d_ws, size_t ws_size,
                              hipStream_t stream) {
    const float* x     = (const float*)d_in[0];
    const float* ln1_g = (const float*)d_in[1];
    const float* ln1_b = (const float*)d_in[2];
    const float* qkv_w = (const float*)d_in[3];
    const float* qkv_b = (const float*)d_in[4];
    const float* out_w = (const float*)d_in[5];
    const float* out_b = (const float*)d_in[6];
    const float* ln2_g = (const float*)d_in[7];
    const float* ln2_b = (const float*)d_in[8];
    const float* w1    = (const float*)d_in[9];
    const float* b1    = (const float*)d_in[10];
    const float* w2    = (const float*)d_in[11];
    const float* b2    = (const float*)d_in[12];
    float* out = (float*)d_out;

    char* ws = (char*)d_ws;
    size_t off = 0;
    auto take = [&](size_t bytes) { void* p = ws + off; off += (bytes + 255) & ~(size_t)255; return p; };

    __bf16* qkvw_b  = (__bf16*)take((size_t)DD * 3 * DD * 2);
    __bf16* outw_b  = (__bf16*)take((size_t)DD * DD * 2);
    __bf16* w1_b    = (__bf16*)take((size_t)DD * FFD * 2);
    __bf16* w2_b    = (__bf16*)take((size_t)FFD * DD * 2);
    __bf16* h1_b    = (__bf16*)take((size_t)MM * DD * 2);
    __bf16* qkv_b16 = (__bf16*)take((size_t)MM * 3 * DD * 2);
    __bf16* attn_b  = (__bf16*)take((size_t)MM * DD * 2);
    float*  x1_f    = (float*) take((size_t)MM * DD * 4);
    __bf16* h2_b    = (__bf16*)take((size_t)MM * DD * 2);
    __bf16* mid_b   = (__bf16*)take((size_t)MM * FFD * 2);

    auto cvt = [&](const float* src, __bf16* dst, size_t n) {
        size_t n4 = n / 4;
        cvt_f32_bf16_kernel<<<(unsigned)((n4 + 255) / 256), 256, 0, stream>>>(src, dst, n4);
    };
    cvt(qkv_w, qkvw_b, (size_t)DD * 3 * DD);
    cvt(out_w, outw_b, (size_t)DD * DD);
    cvt(w1,    w1_b,   (size_t)DD * FFD);
    cvt(w2,    w2_b,   (size_t)FFD * DD);

    // LN1
    layernorm_bf16_kernel<<<MM, 256, 0, stream>>>(x, ln1_g, ln1_b, h1_b);

    // qkv = h1 @ qkv_w + qkv_b  -> bf16
    gemm_bf16_kernel<true, false, false, false, true>
        <<<dim3(3 * DD / 128, MM / 128), 256, 0, stream>>>(
            h1_b, qkvw_b, qkv_b, nullptr, nullptr, qkv_b16, MM, 3 * DD, DD);

    // flash attention -> attn_b (bf16)
    flash_attn_kernel<<<dim3(SS / 64, BB * HH), 128, 0, stream>>>(qkv_b16, attn_b);

    // x1 = x + attn @ out_w + out_b  -> fp32
    gemm_bf16_kernel<true, false, true, true, false>
        <<<dim3(DD / 128, MM / 128), 256, 0, stream>>>(
            attn_b, outw_b, out_b, x, x1_f, nullptr, MM, DD, DD);

    // LN2
    layernorm_bf16_kernel<<<MM, 256, 0, stream>>>(x1_f, ln2_g, ln2_b, h2_b);

    // mid = gelu(h2 @ w1 + b1) -> bf16
    gemm_bf16_kernel<true, true, false, false, true>
        <<<dim3(FFD / 128, MM / 128), 256, 0, stream>>>(
            h2_b, w1_b, b1, nullptr, nullptr, mid_b, MM, FFD, DD);

    // out = x1 + mid @ w2 + b2 -> fp32
    gemm_bf16_kernel<true, false, true, true, false>
        <<<dim3(DD / 128, MM / 128), 256, 0, stream>>>(
            mid_b, w2_b, b2, x1_f, out, nullptr, MM, DD, FFD);
}
